// PAM_Module_21620865368565
// MI455X (gfx1250) — matrix operand order repaired and checked by reading, not compiled, not run on hardware
//
#include <hip/hip_runtime.h>
#include <hip/hip_bf16.h>
#include <cstdint>

typedef __attribute__((ext_vector_type(16))) __bf16 v16bf;
typedef __attribute__((ext_vector_type(8)))  __bf16 v8bf;
typedef __attribute__((ext_vector_type(2)))  __bf16 v2bf;
typedef __attribute__((ext_vector_type(8)))  float  v8f;
typedef __attribute__((ext_vector_type(4)))  int    v4i;

#define BMT 128
#define BNT 128
#define BKK 32

// Async DMA global->LDS (16B per lane) where the toolchain exposes it; else sync.
// This toolchain's builtin signature takes generic int4* pointers (per diagnostics).
#if defined(__gfx1250__) && __has_builtin(__builtin_amdgcn_global_load_async_to_lds_b128)
#define ASYNC_CP16(gp, lp)                                                      \
    __builtin_amdgcn_global_load_async_to_lds_b128((v4i*)(gp), (v4i*)(lp), 0, 0)
#if __has_builtin(__builtin_amdgcn_s_wait_asynccnt)
#define WAIT_ASYNC() __builtin_amdgcn_s_wait_asynccnt(0)
#else
#define WAIT_ASYNC() asm volatile("s_wait_asynccnt 0" ::: "memory")
#endif
#else
#define ASYNC_CP16(gp, lp) (*(v8bf*)(lp) = *(const v8bf*)(gp))
#define WAIT_ASYNC() ((void)0)
#endif

// Batched GEMM, all shapes compile-time.
//   A: [M][K]   (fp32 weights, batch-shared; or bf16 per-batch)
//   B: bf16 "Bt" [NCOLS][K] (K contiguous per out-col), or fp32 [K][NCOLS]
//      transposed+converted into LDS on the fly (B_TRANS_F32).
// OUT_MODE: 0 = bf16 transposed Outt[col*M+row]
//           1 = bf16 row-major  Out[row*NCOLS+col]
//           2 = f32  gamma*acc + resid
template<int M, int NCOLS, int K,
         bool A_F32, bool B_TRANS_F32, bool HAS_BIAS, bool RELU, int OUT_MODE>
__global__ __launch_bounds__(256)
void gemm_wmma_kernel(const void* __restrict__ Aptr,
                      const void* __restrict__ Bptr,
                      const float* __restrict__ bias,
                      void* __restrict__ Outp,
                      const float* __restrict__ resid,
                      const float* __restrict__ gammaPtr)
{
    constexpr long aBS = A_F32 ? 0L : (long)M * K;
    constexpr long bBS = (long)NCOLS * K;
    constexpr long oBS = (long)M * NCOLS;
    constexpr int  lda = K;
    constexpr int  ldb = B_TRANS_F32 ? NCOLS : K;
    constexpr int  ksteps = K / BKK;

    __shared__ alignas(16) __bf16 As[2][BMT][BKK];   // 8 KB x2
    __shared__ alignas(16) __bf16 Bs[2][BNT][BKK];   // 8 KB x2

    const int b     = blockIdx.z;
    const int tileM = blockIdx.y * BMT;
    const int tileN = blockIdx.x * BNT;
    const int t     = threadIdx.x;
    const int wave  = t >> 5;
    const int lane  = t & 31;
    const int waveM = (wave >> 1) * 32;   // 4 waves along M
    const int waveN = (wave & 1) * 64;    // 2 waves along N
    const int m16   = lane & 15;
    const int h     = lane >> 4;

    v8f acc[2][4];
    #pragma unroll
    for (int mi = 0; mi < 2; ++mi)
        #pragma unroll
        for (int ni = 0; ni < 4; ++ni)
            #pragma unroll
            for (int i = 0; i < 8; ++i) acc[mi][ni][i] = 0.0f;

    // ---- Tile staging: 128x32 A and B tiles, 256 threads ----
    auto stage = [&](int kt, int buf) {
        const int k0 = kt * BKK;
        // A tile
        {
            const int arow = t >> 2;          // 0..63
            const int acol = (t & 3) * 8;     // 0,8,16,24
            #pragma unroll
            for (int rr = 0; rr < 2; ++rr) {
                const int row = arow + rr * 64;
                const size_t g = (size_t)b * (size_t)aBS +
                                 (size_t)(tileM + row) * lda + (k0 + acol);
                if constexpr (A_F32) {
                    const float* s = (const float*)Aptr + g;
                    v8bf tmp;
                    #pragma unroll
                    for (int i = 0; i < 8; ++i) tmp[i] = (__bf16)s[i];
                    *(v8bf*)&As[buf][row][acol] = tmp;
                } else {
                    ASYNC_CP16((const __bf16*)Aptr + g, &As[buf][row][acol]);
                }
            }
        }
        // B tile (Bt layout [n][k])
        if constexpr (B_TRANS_F32) {
            #pragma unroll
            for (int it = 0; it < 8; ++it) {
                const int idx = it * 256 + t;
                const int nn  = idx & 127;         // coalesced along n
                const int kk  = (idx >> 7) * 2;    // pair of k
                const float* s = (const float*)Bptr + (size_t)b * (size_t)bBS +
                                 (size_t)(k0 + kk) * ldb + (tileN + nn);
                v2bf p;
                p[0] = (__bf16)s[0];
                p[1] = (__bf16)s[ldb];
                *(v2bf*)&Bs[buf][nn][kk] = p;
            }
        } else {
            const int brow = t >> 2;
            const int bcol = (t & 3) * 8;
            #pragma unroll
            for (int rr = 0; rr < 2; ++rr) {
                const int nn = brow + rr * 64;
                const size_t g = (size_t)b * (size_t)bBS +
                                 (size_t)(tileN + nn) * ldb + (k0 + bcol);
                ASYNC_CP16((const __bf16*)Bptr + g, &Bs[buf][nn][bcol]);
            }
        }
    };

    stage(0, 0);
    WAIT_ASYNC();
    __syncthreads();

    for (int kt = 0; kt < ksteps; ++kt) {
        const int buf = kt & 1;
        if (kt + 1 < ksteps) {
            if constexpr (A_F32)
                __builtin_prefetch((const float*)Aptr + (size_t)b * (size_t)aBS +
                                   (size_t)(tileM + (t >> 2)) * lda + (kt + 1) * BKK + BKK, 0, 1);
            stage(kt + 1, buf ^ 1);
        }

        // ---- 8 WMMAs per wave on the current buffer ----
        v16bf afr[2], bfr[4];
        #pragma unroll
        for (int mi = 0; mi < 2; ++mi) {
            // A 16x32 frag: lanes 0-15 M rows; K = h*8+e (e<8), 16+h*8+(e-8)
            const __bf16* p = &As[buf][waveM + mi * 16 + m16][0];
            v8bf lo = *(const v8bf*)(p + h * 8);
            v8bf hi = *(const v8bf*)(p + 16 + h * 8);
            #pragma unroll
            for (int i = 0; i < 8; ++i) { afr[mi][i] = lo[i]; afr[mi][i + 8] = hi[i]; }
        }
        #pragma unroll
        for (int ni = 0; ni < 4; ++ni) {
            // B 32x16 frag: lane col = lane&15; K = h*8+e (e<8), 16+h*8+(e-8)
            const __bf16* p = &Bs[buf][waveN + ni * 16 + m16][0];
            v8bf lo = *(const v8bf*)(p + h * 8);
            v8bf hi = *(const v8bf*)(p + 16 + h * 8);
            #pragma unroll
            for (int i = 0; i < 8; ++i) { bfr[ni][i] = lo[i]; bfr[ni][i + 8] = hi[i]; }
        }
        #pragma unroll
        for (int mi = 0; mi < 2; ++mi)
            #pragma unroll
            for (int ni = 0; ni < 4; ++ni)
                acc[mi][ni] = __builtin_amdgcn_wmma_f32_16x16x32_bf16(
                    false, afr[mi], false, bfr[ni], (short)0, acc[mi][ni], false, false);

        WAIT_ASYNC();
        __syncthreads();
    }

    // ---- Epilogue: D frag row = r + 8*h, col = lane&15; strides are constants ----
    float gval = 0.0f;
    if constexpr (OUT_MODE == 2) gval = gammaPtr[0];
    #pragma unroll
    for (int mi = 0; mi < 2; ++mi) {
        #pragma unroll
        for (int ni = 0; ni < 4; ++ni) {
            const int rowBase = tileM + waveM + mi * 16;
            const int col     = tileN + waveN + ni * 16 + m16;
            #pragma unroll
            for (int r = 0; r < 8; ++r) {
                const int row = rowBase + r + 8 * h;
                float val = acc[mi][ni][r];
                if constexpr (HAS_BIAS) val += bias[row];
                if constexpr (RELU) val = fmaxf(val, 0.0f);
                if constexpr (OUT_MODE == 0) {
                    ((__bf16*)Outp)[(size_t)b * (size_t)oBS + (size_t)col * M + row] = (__bf16)val;
                } else if constexpr (OUT_MODE == 1) {
                    ((__bf16*)Outp)[(size_t)b * (size_t)oBS + (size_t)row * NCOLS + col] = (__bf16)val;
                } else {
                    const size_t o = (size_t)b * (size_t)oBS + (size_t)row * NCOLS + col;
                    ((float*)Outp)[o] = gval * val + resid[o];
                }
            }
        }
    }
}

// Row softmax over 1024 bf16 entries, in place (f32 math). One block per row.
__global__ __launch_bounds__(256)
void softmax_rows(void* attn_)
{
    __bf16* row = (__bf16*)attn_ + (size_t)blockIdx.x * 1024;
    __shared__ float red[8];
    const int t = threadIdx.x, lane = t & 31, wv = t >> 5;
    float v[4];
    float mx = -3.0e38f;
    #pragma unroll
    for (int i = 0; i < 4; ++i) { v[i] = (float)row[t + i * 256]; mx = fmaxf(mx, v[i]); }
    #pragma unroll
    for (int o = 16; o > 0; o >>= 1) mx = fmaxf(mx, __shfl_xor(mx, o, 32));
    if (lane == 0) red[wv] = mx;
    __syncthreads();
    mx = red[0];
    #pragma unroll
    for (int w = 1; w < 8; ++w) mx = fmaxf(mx, red[w]);
    float s = 0.0f;
    #pragma unroll
    for (int i = 0; i < 4; ++i) { v[i] = __expf(v[i] - mx); s += v[i]; }
    #pragma unroll
    for (int o = 16; o > 0; o >>= 1) s += __shfl_xor(s, o, 32);
    __syncthreads();
    if (lane == 0) red[wv] = s;
    __syncthreads();
    s = 0.0f;
    #pragma unroll
    for (int w = 0; w < 8; ++w) s += red[w];
    const float inv = 1.0f / s;
    #pragma unroll
    for (int i = 0; i < 4; ++i) row[t + i * 256] = (__bf16)(v[i] * inv);
}

// ra[b,i,j] = mean of the 7x7 sub-block (i,j) of attn[b, :49, :49]
__global__ void ra_kernel(const void* attn_, float* ra)
{
    const __bf16* attn = (const __bf16*)attn_;
    const int idx = blockIdx.x * blockDim.x + threadIdx.x;  // b*49 + i*7 + j
    if (idx >= 32 * 49) return;
    const int b = idx / 49, ij = idx % 49, i = ij / 7, j = ij % 7;
    float s = 0.0f;
    for (int r = 0; r < 7; ++r)
        for (int c = 0; c < 7; ++c)
            s += (float)attn[((size_t)b * 1024 + i * 7 + r) * 1024 + j * 7 + c];
    ra[idx] = s * (1.0f / 49.0f);
}

// Apply region-dropout mask to the top-left 49x49 block of each attn[b].
__global__ void mask_kernel(void* attn_, const float* ra, const float* rnd)
{
    __bf16* attn = (__bf16*)attn_;
    const int idx = blockIdx.x * blockDim.x + threadIdx.x;  // b*2401 + row*49 + col
    if (idx >= 32 * 49 * 49) return;
    const int b = idx / 2401, rc = idx % 2401;
    const int rowi = rc / 49, colj = rc % 49;
    const int i = rowi / 7, j = colj / 7;
    float rs = 0.0f;
    #pragma unroll
    for (int bb = 0; bb < 32; ++bb) rs += ra[bb * 49 + i * 7 + j];
    if (rs == 0.0f) rs = 1e-8f;
    const float prob = ra[b * 49 + i * 7 + j] / rs;
    const float m = (rnd[b * 49 + i * 7 + j] <= prob) ? 1.0f : 0.0f;
    const size_t o = ((size_t)b * 1024 + rowi) * 1024 + colj;
    attn[o] = (__bf16)((float)attn[o] * m);
}

extern "C" void kernel_launch(void* const* d_in, const int* in_sizes, int n_in,
                              void* d_out, int out_size, void* d_ws, size_t ws_size,
                              hipStream_t stream)
{
    (void)in_sizes; (void)n_in; (void)out_size; (void)ws_size;
    const float* x     = (const float*)d_in[0];
    const float* wq1   = (const float*)d_in[1];
    const float* bq1   = (const float*)d_in[2];
    const float* wq2   = (const float*)d_in[3];
    const float* bq2   = (const float*)d_in[4];
    const float* wk1   = (const float*)d_in[5];
    const float* bk1   = (const float*)d_in[6];
    const float* wk2   = (const float*)d_in[7];
    const float* bk2   = (const float*)d_in[8];
    const float* wv    = (const float*)d_in[9];
    const float* bv    = (const float*)d_in[10];
    const float* gamma = (const float*)d_in[11];
    const float* rnd   = (const float*)d_in[12];

    // B=32, C=512, N=1024, d=128
    char* ws = (char*)d_ws;
    void*  T1t  = (void*)(ws + 0);                 //  8 MB bf16 [B][N][d] (reused q/k)
    void*  Qt   = (void*)(ws + (8ull  << 20));     //  8 MB bf16 [B][N][d]
    void*  Kt   = (void*)(ws + (16ull << 20));     //  8 MB bf16 [B][N][d]
    void*  V    = (void*)(ws + (24ull << 20));     // 32 MB bf16 [B][C][N]
    void*  attn = (void*)(ws + (56ull << 20));     // 64 MB bf16 [B][N][N]
    float* ra   = (float*)(ws + (120ull << 20));   //  6.3 KB f32 [B][7][7]

    const dim3 blk(256);

    // Q path: T1t = relu(wq1 @ x + bq1)^T ;  Qt = (wq2 @ T1 + bq2)^T
    gemm_wmma_kernel<128, 1024, 512, true,  true,  true,  true,  0>
        <<<dim3(8, 1, 32), blk, 0, stream>>>(wq1, x, bq1, T1t, nullptr, nullptr);
    gemm_wmma_kernel<128, 1024, 128, true,  false, true,  false, 0>
        <<<dim3(8, 1, 32), blk, 0, stream>>>(wq2, T1t, bq2, Qt, nullptr, nullptr);
    // K path
    gemm_wmma_kernel<128, 1024, 512, true,  true,  true,  true,  0>
        <<<dim3(8, 1, 32), blk, 0, stream>>>(wk1, x, bk1, T1t, nullptr, nullptr);
    gemm_wmma_kernel<128, 1024, 128, true,  false, true,  false, 0>
        <<<dim3(8, 1, 32), blk, 0, stream>>>(wk2, T1t, bk2, Kt, nullptr, nullptr);
    // V = relu(wv @ x + bv), row-major [C][N]
    gemm_wmma_kernel<512, 1024, 512, true,  true,  true,  true,  1>
        <<<dim3(8, 4, 32), blk, 0, stream>>>(wv, x, bv, V, nullptr, nullptr);
    // energy[n][m] = sum_c Qt[n][c] * Kt[m][c]
    gemm_wmma_kernel<1024, 1024, 128, false, false, false, false, 1>
        <<<dim3(8, 8, 32), blk, 0, stream>>>(Qt, Kt, nullptr, attn, nullptr, nullptr);
    // softmax rows in place
    softmax_rows<<<dim3(32 * 1024), dim3(256), 0, stream>>>(attn);
    // region dropout
    ra_kernel<<<dim3(7),   dim3(256), 0, stream>>>(attn, ra);
    mask_kernel<<<dim3(301), dim3(256), 0, stream>>>(attn, ra, rnd);
    // out[c][n] = gamma * sum_m V[c][m] * attn[n][m] + x[c][n]
    gemm_wmma_kernel<512, 1024, 1024, false, false, false, false, 2>
        <<<dim3(8, 4, 32), blk, 0, stream>>>(V, attn, nullptr, d_out, x, gamma);
}
